// Attention_2104533975416
// MI455X (gfx1250) — matrix operand order repaired and checked by reading, not compiled, not run on hardware
//
#include <hip/hip_runtime.h>

#define DIM 1024
#define HEADS 16
#define DIM_HEAD 64
#define SEQ 2048
#define BATCH 2
#define TOKENS (BATCH * SEQ)  // 4096

typedef __attribute__((ext_vector_type(16))) __bf16 v16bf;
typedef __attribute__((ext_vector_type(8)))  float  v8f;
typedef __attribute__((ext_vector_type(4)))  unsigned int u32x4;
typedef __attribute__((ext_vector_type(8)))  int i32x8;
typedef __attribute__((ext_vector_type(4)))  int i32x4;

union Frag {
  v16bf v;
  u32x4 q[2];
};

__device__ __forceinline__ unsigned short f32_to_bf16_rne(float f) {
  unsigned int u = __builtin_bit_cast(unsigned int, f);
  u += 0x7fffu + ((u >> 16) & 1u);
  return (unsigned short)(u >> 16);
}

__device__ __forceinline__ v8f wmma_bf16(const Frag& a, const Frag& b, v8f c) {
  return __builtin_amdgcn_wmma_f32_16x16x32_bf16(false, a.v, false, b.v,
                                                 (short)0, c, false, false);
}

// ---------------------------------------------------------------------------
// TDM: issue a 2D tile load (32 elems x 128 rows, bf16) global -> LDS with
// row padding 4 dwords per 16 dwords (=> LDS row stride 40 ushorts).
// Descriptor bit layout per CDNA5 ISA ch.8 (D# groups 0/1; groups 2/3 zero).
// ---------------------------------------------------------------------------
__device__ __forceinline__ void tdm_load_tile_2d(unsigned lds_addr,
                                                 const unsigned short* gtile,
                                                 unsigned rem_k, unsigned rem_rows,
                                                 unsigned lda_elems) {
  unsigned long long ga = (unsigned long long)(uintptr_t)gtile;
  u32x4 g0;
  g0[0] = 1u;                                   // count=1, user descriptor
  g0[1] = lds_addr;                             // lds_addr (bytes)
  g0[2] = (unsigned)(ga & 0xFFFFFFFFu);         // global_addr[31:0]
  g0[3] = (unsigned)(ga >> 32) | 0x80000000u;   // global_addr[56:32] | type=2

  union { i32x8 v; unsigned long long q[4]; } g1;
  // dw0: data_size=1 (2B) <<16 | pad_enable<<20 | pad_interval=3 (16 dw) <<22
  //      | pad_amount=3 (4 dw) <<25 ; workgroup_mask=0
  unsigned long long flags = (1ull << 16) | (1ull << 20) | (3ull << 22) | (3ull << 25);
  unsigned long long td0 = rem_k;          // tensor_dim0  [79:48]
  unsigned long long td1 = rem_rows;       // tensor_dim1  [111:80]
  unsigned long long t0  = 32;             // tile_dim0    [127:112]
  unsigned long long t1  = 128;            // tile_dim1    [143:128]
  unsigned long long t2  = 0;              // tile_dim2    [159:144]
  unsigned long long st0 = lda_elems;      // tensor_dim0_stride [207:160]
  unsigned long long st1 = 0;              // tensor_dim1_stride [255:208]
  g1.q[0] = flags | ((td0 & 0xFFFFull) << 48);
  g1.q[1] = (td0 >> 16) | (td1 << 16) | (t0 << 48);
  g1.q[2] = t1 | (t2 << 16) | ((st0 & 0xFFFFFFFFull) << 32);
  g1.q[3] = (st0 >> 32) | (st1 << 16);

  i32x4 z4 = {0, 0, 0, 0};
#if defined(__clang_major__) && (__clang_major__ >= 23)
  i32x8 z8 = {0, 0, 0, 0, 0, 0, 0, 0};
  __builtin_amdgcn_tensor_load_to_lds(g0, g1.v, z4, z4, z8, 0);
#else
  __builtin_amdgcn_tensor_load_to_lds(g0, g1.v, z4, z4, 0);
#endif
}

// ---------------------------------------------------------------------------
// Convert f32 weight [K][N] -> bf16 transposed [N][K]
// ---------------------------------------------------------------------------
__global__ void k_cvt_transpose(const float* __restrict__ W,
                                unsigned short* __restrict__ Wt, int K, int N) {
  int idx = blockIdx.x * blockDim.x + threadIdx.x;
  if (idx >= K * N) return;
  int k = idx / N, n = idx % N;
  Wt[(size_t)n * K + k] = f32_to_bf16_rne(W[idx]);
}

// ---------------------------------------------------------------------------
// RMSNorm (+bf16 x-hat) + sigmoid gates, one block per token row
// ---------------------------------------------------------------------------
__global__ __launch_bounds__(256) void k_rmsnorm_gates(
    const float* __restrict__ tokens, const float* __restrict__ norm_w,
    const float* __restrict__ Wg, unsigned short* __restrict__ xn,
    float* __restrict__ gates) {
  __shared__ float xs[DIM];
  __shared__ float red[256];
  __shared__ float gp[16][17];
  const int row = blockIdx.x, t = threadIdx.x;
  const float* x = tokens + (size_t)row * DIM;

  float v[4];
  float ss = 0.f;
#pragma unroll
  for (int j = 0; j < 4; ++j) {
    v[j] = x[t + 256 * j];
    ss += v[j] * v[j];
  }
  red[t] = ss;
  __syncthreads();
  for (int s2 = 128; s2 > 0; s2 >>= 1) {
    if (t < s2) red[t] += red[t + s2];
    __syncthreads();
  }
  const float r = rsqrtf(red[0] * (1.f / DIM) + 1.1920929e-7f);
#pragma unroll
  for (int j = 0; j < 4; ++j) {
    int c = t + 256 * j;
    float xv = v[j] * r * norm_w[c];
    xs[c] = xv;
    xn[(size_t)row * DIM + c] = f32_to_bf16_rne(xv);
  }
  __syncthreads();
  int h = t & 15, chunk = t >> 4;
  float acc = 0.f;
  for (int c = chunk * 64; c < chunk * 64 + 64; ++c) acc += xs[c] * Wg[c * HEADS + h];
  gp[chunk][h] = acc;
  __syncthreads();
  if (t < 16) {
    float s = 0.f;
#pragma unroll
    for (int c2 = 0; c2 < 16; ++c2) s += gp[c2][t];
    gates[(size_t)row * HEADS + t] = 1.f / (1.f + __expf(-s));
  }
}

// ---------------------------------------------------------------------------
// bf16 GEMM: C[M,N] = A[M,K] * Bt[N,K]^T, 128x128 block tile, BK=32.
// Tile staging via Tensor Data Mover, double-buffered (wave 0 issues,
// TENSORcnt gates, block barrier publishes).
// mode 0: f32 store C[row*N+col]
// mode 1: bf16 scatter to [b,h,n,d]  (q/k buffers)
// mode 2: bf16 scatter to [b,h,d,n]  (v transposed)
// ---------------------------------------------------------------------------
__global__ __launch_bounds__(256) void k_gemm_bf16(
    const unsigned short* __restrict__ A, int lda,
    const unsigned short* __restrict__ Bt, int ldb, int M, int N, int K,
    int mode, void* __restrict__ outp) {
  __shared__ __align__(16) unsigned short As[2][128][40];
  __shared__ __align__(16) unsigned short Bs[2][128][40];
  const int t = threadIdx.x;
  const int bm = blockIdx.y * 128, bn = blockIdx.x * 128;
  const int wid = t >> 5, lane = t & 31, lr = lane & 15, lh = lane >> 4;
  const int wm = (wid & 3) * 32, wn = (wid >> 2) * 64;
  const bool issuer = (wid == 0);

  v8f acc[2][4];
#pragma unroll
  for (int i = 0; i < 2; ++i)
#pragma unroll
    for (int j = 0; j < 4; ++j)
#pragma unroll
      for (int e = 0; e < 8; ++e) acc[i][j][e] = 0.f;

  const unsigned ldsA[2] = {(unsigned)(uintptr_t)&As[0][0][0],
                            (unsigned)(uintptr_t)&As[1][0][0]};
  const unsigned ldsB[2] = {(unsigned)(uintptr_t)&Bs[0][0][0],
                            (unsigned)(uintptr_t)&Bs[1][0][0]};

  const int nsteps = K / 32;
  if (issuer) {
    tdm_load_tile_2d(ldsA[0], A + (size_t)bm * lda, K, M - bm, lda);
    tdm_load_tile_2d(ldsB[0], Bt + (size_t)bn * ldb, K, N - bn, ldb);
  }

  for (int it = 0; it < nsteps; ++it) {
    const int cur = it & 1;
    if (issuer) {
      if (it + 1 < nsteps) {
        const int k1 = (it + 1) * 32;
        tdm_load_tile_2d(ldsA[cur ^ 1], A + (size_t)bm * lda + k1, K - k1, M - bm, lda);
        tdm_load_tile_2d(ldsB[cur ^ 1], Bt + (size_t)bn * ldb + k1, K - k1, N - bn, ldb);
        __builtin_amdgcn_s_wait_tensorcnt((short)2);  // tile `it` landed
      } else {
        __builtin_amdgcn_s_wait_tensorcnt((short)0);
      }
    }
    __syncthreads();  // publish tile `it` to all waves

    Frag af[2], bfr[4];
#pragma unroll
    for (int mf = 0; mf < 2; ++mf) {
      af[mf].q[0] = *(const u32x4*)&As[cur][wm + mf * 16 + lr][lh * 8];
      af[mf].q[1] = *(const u32x4*)&As[cur][wm + mf * 16 + lr][16 + lh * 8];
    }
#pragma unroll
    for (int nf = 0; nf < 4; ++nf) {
      bfr[nf].q[0] = *(const u32x4*)&Bs[cur][wn + nf * 16 + lr][lh * 8];
      bfr[nf].q[1] = *(const u32x4*)&Bs[cur][wn + nf * 16 + lr][16 + lh * 8];
    }
#pragma unroll
    for (int mf = 0; mf < 2; ++mf)
#pragma unroll
      for (int nf = 0; nf < 4; ++nf)
        acc[mf][nf] = wmma_bf16(af[mf], bfr[nf], acc[mf][nf]);

    __syncthreads();  // all waves done with buffer `cur`; safe to overwrite at it+2
  }

#pragma unroll
  for (int mf = 0; mf < 2; ++mf) {
#pragma unroll
    for (int nf = 0; nf < 4; ++nf) {
      int col = bn + wn + nf * 16 + lr;
#pragma unroll
      for (int e = 0; e < 8; ++e) {
        int row = bm + wm + mf * 16 + e + 8 * lh;
        float val = acc[mf][nf][e];
        if (mode == 0) {
          ((float*)outp)[(size_t)row * N + col] = val;
        } else {
          int b = row >> 11, n = row & (SEQ - 1);
          int hh = col >> 6, d = col & 63;
          size_t idx;
          if (mode == 1)
            idx = (((size_t)(b * HEADS + hh)) * SEQ + n) * DIM_HEAD + d;
          else
            idx = (((size_t)(b * HEADS + hh)) * DIM_HEAD + d) * SEQ + n;
          ((unsigned short*)outp)[idx] = f32_to_bf16_rne(val);
        }
      }
    }
  }
}

// ---------------------------------------------------------------------------
// Causal flash attention + gating. Each wave owns 16 query rows.
// q/k: [b,h,n,d] bf16; v: [b,h,d,n] bf16; out ao: [b,n,h*d] bf16
// ---------------------------------------------------------------------------
__global__ __launch_bounds__(256) void k_flash(
    const unsigned short* __restrict__ qb, const unsigned short* __restrict__ kb,
    const unsigned short* __restrict__ vt, const float* __restrict__ gates,
    unsigned short* __restrict__ ao) {
  __shared__ __align__(16) unsigned short pbuf[8][16][72];  // per-wave P tile
  const int t = threadIdx.x, wid = t >> 5, lane = t & 31;
  const int lr = lane & 15, lh = lane >> 4;
  const int bh = blockIdx.y, b = bh >> 4, h = bh & 15;
  const int qtb = blockIdx.x * 128 + wid * 16;
  const size_t base = (size_t)bh * SEQ * DIM_HEAD;
  const size_t vbase = (size_t)bh * DIM_HEAD * SEQ;

  Frag aq[2];
  {
    const unsigned short* qr = qb + base + (size_t)(qtb + lr) * DIM_HEAD;
#pragma unroll
    for (int c = 0; c < 2; ++c) {
      aq[c].q[0] = *(const u32x4*)(qr + c * 32 + lh * 8);
      aq[c].q[1] = *(const u32x4*)(qr + c * 32 + 16 + lh * 8);
    }
  }

  float mrow[8], lrow[8];
  v8f o[4];
#pragma unroll
  for (int e = 0; e < 8; ++e) { mrow[e] = -3.0e38f; lrow[e] = 0.f; }
#pragma unroll
  for (int dt = 0; dt < 4; ++dt)
#pragma unroll
    for (int e = 0; e < 8; ++e) o[dt][e] = 0.f;

  const float scale = 0.125f;  // 64^-0.5
  for (int jb = 0; jb < qtb + 16; jb += 64) {
    v8f s[4];
#pragma unroll
    for (int jt = 0; jt < 4; ++jt) {
#pragma unroll
      for (int e = 0; e < 8; ++e) s[jt][e] = 0.f;
      const unsigned short* kr = kb + base + (size_t)(jb + jt * 16 + lr) * DIM_HEAD;
#pragma unroll
      for (int c = 0; c < 2; ++c) {
        Frag bk;
        bk.q[0] = *(const u32x4*)(kr + c * 32 + lh * 8);
        bk.q[1] = *(const u32x4*)(kr + c * 32 + 16 + lh * 8);
        s[jt] = wmma_bf16(aq[c], bk, s[jt]);
      }
    }
#pragma unroll
    for (int jt = 0; jt < 4; ++jt) {
      int j = jb + jt * 16 + lr;
#pragma unroll
      for (int e = 0; e < 8; ++e) {
        int row = qtb + e + 8 * lh;
        float sv = s[jt][e] * scale;
        s[jt][e] = (j > row) ? -3.0e38f : sv;
      }
    }
    float rs[8], alpha[8];
#pragma unroll
    for (int e = 0; e < 8; ++e) {
      float mx = s[0][e];
#pragma unroll
      for (int jt = 1; jt < 4; ++jt) mx = fmaxf(mx, s[jt][e]);
#pragma unroll
      for (int off = 8; off >= 1; off >>= 1) mx = fmaxf(mx, __shfl_xor(mx, off, 32));
      float mn = fmaxf(mrow[e], mx);
      alpha[e] = __expf(mrow[e] - mn);
      mrow[e] = mn;
      rs[e] = 0.f;
    }
#pragma unroll
    for (int jt = 0; jt < 4; ++jt)
#pragma unroll
      for (int e = 0; e < 8; ++e) {
        float p = __expf(s[jt][e] - mrow[e]);
        s[jt][e] = p;
        rs[e] += p;
      }
#pragma unroll
    for (int e = 0; e < 8; ++e) {
#pragma unroll
      for (int off = 8; off >= 1; off >>= 1) rs[e] += __shfl_xor(rs[e], off, 32);
      lrow[e] = lrow[e] * alpha[e] + rs[e];
    }
#pragma unroll
    for (int dt = 0; dt < 4; ++dt)
#pragma unroll
      for (int e = 0; e < 8; ++e) o[dt][e] *= alpha[e];

    // P (C layout f32) -> bf16 LDS -> A-fragment layout; wave-private region,
    // intra-wave ordering via split DS counter.
#pragma unroll
    for (int jt = 0; jt < 4; ++jt)
#pragma unroll
      for (int e = 0; e < 8; ++e)
        pbuf[wid][e + 8 * lh][jt * 16 + lr] = f32_to_bf16_rne(s[jt][e]);
    asm volatile("s_wait_dscnt 0" ::: "memory");
    Frag ap[2];
#pragma unroll
    for (int c = 0; c < 2; ++c) {
      ap[c].q[0] = *(const u32x4*)&pbuf[wid][lr][c * 32 + lh * 8];
      ap[c].q[1] = *(const u32x4*)&pbuf[wid][lr][c * 32 + 16 + lh * 8];
    }
#pragma unroll
    for (int dt = 0; dt < 4; ++dt) {
      const unsigned short* vr = vt + vbase + (size_t)(dt * 16 + lr) * SEQ + jb;
#pragma unroll
      for (int c = 0; c < 2; ++c) {
        Frag bv;
        bv.q[0] = *(const u32x4*)(vr + c * 32 + lh * 8);
        bv.q[1] = *(const u32x4*)(vr + c * 32 + 16 + lh * 8);
        o[dt] = wmma_bf16(ap[c], bv, o[dt]);
      }
    }
  }

  float g[8];
#pragma unroll
  for (int e = 0; e < 8; ++e) {
    int row = qtb + e + 8 * lh;
    g[e] = gates[(size_t)(b * SEQ + row) * HEADS + h] / lrow[e];
  }
#pragma unroll
  for (int dt = 0; dt < 4; ++dt) {
#pragma unroll
    for (int e = 0; e < 8; ++e) {
      int row = qtb + e + 8 * lh;
      size_t idx = (size_t)(b * SEQ + row) * (HEADS * DIM_HEAD) + h * DIM_HEAD +
                   dt * 16 + lr;
      ao[idx] = f32_to_bf16_rne(o[dt][e] * g[e]);
    }
  }
}

// ---------------------------------------------------------------------------
extern "C" void kernel_launch(void* const* d_in, const int* in_sizes, int n_in,
                              void* d_out, int out_size, void* d_ws, size_t ws_size,
                              hipStream_t stream) {
  (void)in_sizes; (void)n_in; (void)out_size; (void)ws_size;
  const float* tokens = (const float*)d_in[0];
  const float* norm_w = (const float*)d_in[1];
  const float* Wq     = (const float*)d_in[2];
  const float* Wkv    = (const float*)d_in[3];
  const float* Wout   = (const float*)d_in[4];
  const float* Wg     = (const float*)d_in[5];

  char* p = (char*)d_ws;
  auto carve = [&](size_t bytes) -> char* {
    char* r = p;
    p += (bytes + 255) & ~(size_t)255;
    return r;
  };
  unsigned short* Wq_t   = (unsigned short*)carve((size_t)DIM * DIM * 2);
  unsigned short* Wkv_t  = (unsigned short*)carve((size_t)2 * DIM * DIM * 2);
  unsigned short* Wout_t = (unsigned short*)carve((size_t)DIM * DIM * 2);
  unsigned short* xn     = (unsigned short*)carve((size_t)TOKENS * DIM * 2);
  float*          gate   = (float*)carve((size_t)TOKENS * HEADS * 4);
  unsigned short* qbuf   = (unsigned short*)carve((size_t)TOKENS * DIM * 2);
  unsigned short* kbuf   = (unsigned short*)carve((size_t)TOKENS * DIM * 2);
  unsigned short* vbuf   = (unsigned short*)carve((size_t)TOKENS * DIM * 2);
  unsigned short* aobuf  = (unsigned short*)carve((size_t)TOKENS * DIM * 2);

  k_cvt_transpose<<<(DIM * DIM) / 256, 256, 0, stream>>>(Wq, Wq_t, DIM, DIM);
  k_cvt_transpose<<<(2 * DIM * DIM) / 256, 256, 0, stream>>>(Wkv, Wkv_t, DIM, 2 * DIM);
  k_cvt_transpose<<<(DIM * DIM) / 256, 256, 0, stream>>>(Wout, Wout_t, DIM, DIM);

  k_rmsnorm_gates<<<TOKENS, 256, 0, stream>>>(tokens, norm_w, Wg, xn, gate);

  dim3 gg(DIM / 128, TOKENS / 128);
  k_gemm_bf16<<<gg, 256, 0, stream>>>(xn, DIM, Wq_t, DIM, TOKENS, DIM, DIM, 1, qbuf);
  k_gemm_bf16<<<gg, 256, 0, stream>>>(xn, DIM, Wkv_t, DIM, TOKENS, DIM, DIM, 1, kbuf);
  k_gemm_bf16<<<gg, 256, 0, stream>>>(xn, DIM, Wkv_t + (size_t)DIM * DIM, DIM,
                                      TOKENS, DIM, DIM, 2, vbuf);

  k_flash<<<dim3(SEQ / 128, BATCH * HEADS), 256, 0, stream>>>(qbuf, kbuf, vbuf,
                                                              gate, aobuf);

  k_gemm_bf16<<<gg, 256, 0, stream>>>(aobuf, DIM, Wout_t, DIM, TOKENS, DIM, DIM,
                                      0, d_out);
}
